// PairElementEnergyScattering_20134806683944
// MI455X (gfx1250) — matrix operand order repaired and checked by reading, not compiled, not run on hardware
//
#include <hip/hip_runtime.h>

// ---- problem constants (from reference) ----
#define Z_EMB   64
#define E_DIM   32
#define HID     128
#define OUT_D   64
#define P_N     4096
#define NE      200
#define NB      13          // ceil(NE/16)

// padded LDS strides (bank-conflict avoidance, 4B alignment for b32 pair loads)
#define EP_STRIDE  132      // f32 elements per ep row
#define W2T_STRIDE 132      // f16 elements per W2^T column
#define W3T_STRIDE 132      // f16 elements per W3^T column
#define H2_STRIDE  132      // f16 elements per h2 row

typedef __attribute__((ext_vector_type(16))) _Float16 v16h;
typedef __attribute__((ext_vector_type(8)))  float    v8f;

// ---- fast silu: avoid IEEE division lowering (v_div_scale/fixup/fmas) ----
// Preferred: silu(x) = hx + hx*tanh(hx), hx = x/2  -> mul + v_tanh_f32 + fma
// Fallback:  x * v_rcp_f32(1 + v_exp_f32(-x))      -> 5 VALU ops, no div chain
__device__ __forceinline__ float silu_f(float x) {
#if __has_builtin(__builtin_amdgcn_tanh_f32)
    const float hx = 0.5f * x;
    return hx + hx * __builtin_amdgcn_tanh_f32(hx);
#elif __has_builtin(__builtin_amdgcn_tanhf)
    const float hx = 0.5f * x;
    return hx + hx * __builtin_amdgcn_tanhf(hx);
#else
    return x * __builtin_amdgcn_rcpf(1.0f + __expf(-x));
#endif
}

// ------------------------------------------------------------------
// Kernel 1a: path_part[p, :] = z_emb[z_j[p]] @ W1[0:64] + z_emb[z_k[p]] @ W1[64:128]
// grid = P, block = 128 (one thread per hidden unit)
// ------------------------------------------------------------------
__global__ __launch_bounds__(128)
void pp_kernel(const int* __restrict__ zj, const int* __restrict__ zk,
               const float* __restrict__ z_emb, const float* __restrict__ W1,
               float* __restrict__ pp) {
    __shared__ float ej[Z_EMB];
    __shared__ float ek[Z_EMB];
    const int p = blockIdx.x;
    const int t = threadIdx.x;
    if (t < Z_EMB) {
        ej[t] = z_emb[zj[p] * Z_EMB + t];
    } else {
        ek[t - Z_EMB] = z_emb[zk[p] * Z_EMB + (t - Z_EMB)];
    }
    __syncthreads();
    float acc = 0.0f;
    #pragma unroll 8
    for (int k = 0; k < Z_EMB; ++k) {
        acc += ej[k] * W1[k * HID + t] + ek[k] * W1[(Z_EMB + k) * HID + t];
    }
    pp[p * HID + t] = acc;
}

// ------------------------------------------------------------------
// Kernel 1b: energy_part[n, :] = e_feat[n] @ W1[128:160]
// grid = NE, block = 128
// ------------------------------------------------------------------
__global__ __launch_bounds__(128)
void ep_kernel(const float* __restrict__ e_feat, const float* __restrict__ W1,
               float* __restrict__ ep) {
    __shared__ float ef[E_DIM];
    const int n = blockIdx.x;
    const int t = threadIdx.x;
    if (t < E_DIM) ef[t] = e_feat[n * E_DIM + t];
    __syncthreads();
    float acc = 0.0f;
    #pragma unroll
    for (int k = 0; k < E_DIM; ++k) {
        acc += ef[k] * W1[(2 * Z_EMB + k) * HID + t];
    }
    ep[n * HID + t] = acc;
}

// ------------------------------------------------------------------
// Main fused kernel: per workgroup = one p; 8 waves each own 16-row n-tiles.
//   h   = silu(ppb[p] + ep[n])            (ppb = pp + b1, folded at staging)
//   h2  = silu(h @ W2 + b2)               (WMMA f16 -> f32, via LDS bounce)
//   out = h2 @ W3 + b3                    (WMMA f16 -> f32, stored f32)
// grid = P, block = 256
// ------------------------------------------------------------------
__global__ __launch_bounds__(256)
void mlp_kernel(const float* __restrict__ pp_g, const float* __restrict__ ep_g,
                const float* __restrict__ W2,   const float* __restrict__ b2,
                const float* __restrict__ W3,   const float* __restrict__ b3,
                const float* __restrict__ b1,   float* __restrict__ out) {
    __shared__ _Float16 w2t[HID * W2T_STRIDE];     // [col][k]  (transposed)
    __shared__ _Float16 w3t[OUT_D * W3T_STRIDE];   // [col][k]  (transposed)
    __shared__ float    b2s[HID];
    __shared__ float    b3s[OUT_D];
    __shared__ float    ppb[HID];                  // pp[p] + b1 (per-WG constant)
    __shared__ float    eps[8][16 * EP_STRIDE];    // per-wave: ep rows + ppb (f32)
    __shared__ _Float16 h2s[8][16 * H2_STRIDE];    // per-wave h2 bounce (f16)

    const int p    = blockIdx.x;
    const int tid  = threadIdx.x;
    const int wave = tid >> 5;
    const int lane = tid & 31;
    const int half = lane >> 4;   // K-half / M-half selector per ISA layouts
    const int lm   = lane & 15;   // row (A) or column (B/C/D) index

    // ---- stage weights (transposed, f32->f16) and biases into LDS ----
    for (int i = tid; i < HID * HID; i += 256) {
        const int k = i >> 7, j = i & 127;             // W2[k][j]
        w2t[j * W2T_STRIDE + k] = (_Float16)W2[i];
    }
    for (int i = tid; i < HID * OUT_D; i += 256) {
        const int k = i >> 6, j = i & 63;              // W3[k][j]
        w3t[j * W3T_STRIDE + k] = (_Float16)W3[i];
    }
    if (tid < HID) {
        b2s[tid] = b2[tid];
        ppb[tid] = pp_g[(size_t)p * HID + tid] + b1[tid];
    }
    if (tid < OUT_D) b3s[tid] = b3[tid];
    __syncthreads();

    for (int pass = 0; pass < 2; ++pass) {
        const int nblock = wave + pass * 8;
        if (nblock >= NB) break;                       // no barriers below: safe
        const int n0 = nblock * 16;

        // ---- stage this wave's 16 rows of (ep[n] + pp[p] + b1) ----
        float* epw = eps[wave];
        for (int i = lane; i < 16 * HID; i += 32) {
            const int r = i >> 7, c = i & 127;
            int n = n0 + r; if (n >= NE) n = NE - 1;
            epw[r * EP_STRIDE + c] = ep_g[(size_t)n * HID + c] + ppb[c];
        }

        // ---- build layer-2 A fragments: h = silu(pre-activation), f16 ----
        // 16-bit A 16x32 layout: lane(0..15) row M=lm, VGPR v<4 -> K=2v+8*half,
        // v>=4 -> K=16+2(v-4)+8*half  (per CDNA5 ISA §7.12.2)
        v16h a1[4];
        #pragma unroll
        for (int c = 0; c < 4; ++c) {
            #pragma unroll
            for (int v = 0; v < 8; ++v) {
                const int kl = (v < 4) ? (2 * v) : (16 + 2 * (v - 4));
                const int k  = 32 * c + kl + 8 * half;
                a1[c][2 * v]     = (_Float16)silu_f(epw[lm * EP_STRIDE + k]);
                a1[c][2 * v + 1] = (_Float16)silu_f(epw[lm * EP_STRIDE + k + 1]);
            }
        }

        // ---- layer 2: h2 = silu(h @ W2 + b2) -> per-wave LDS bounce ----
        _Float16* h2w = h2s[wave];
        #pragma unroll
        for (int j = 0; j < 8; ++j) {                  // 8 N-tiles of 16 (HID=128)
            const int col = 16 * j + lm;
            // batch all 4 K-chunk B fragments first (DS clause, hide dscnt waits)
            v16h b[4];
            #pragma unroll
            for (int c = 0; c < 4; ++c) {
                #pragma unroll
                for (int v = 0; v < 8; ++v) {
                    const int k = 32 * c + 2 * v + (v >= 4 ? 8 : 0) + 8 * half;
                    b[c][2 * v]     = w2t[col * W2T_STRIDE + k];
                    b[c][2 * v + 1] = w2t[col * W2T_STRIDE + k + 1];
                }
            }
            v8f acc = {};
            #pragma unroll
            for (int c = 0; c < 4; ++c) {
                acc = __builtin_amdgcn_wmma_f32_16x16x32_f16(
                          false, a1[c], false, b[c], (short)0, acc, false, false);
            }
            const float bb = b2s[col];
            #pragma unroll
            for (int r = 0; r < 8; ++r) {              // D layout: M = r + 8*half
                const int m = r + 8 * half;
                h2w[m * H2_STRIDE + col] = (_Float16)silu_f(acc[r] + bb);
            }
        }

        // ---- reload h2 in A-fragment layout (lane-crossing via LDS) ----
        v16h a2[4];
        #pragma unroll
        for (int c = 0; c < 4; ++c) {
            #pragma unroll
            for (int v = 0; v < 8; ++v) {
                const int kl = (v < 4) ? (2 * v) : (16 + 2 * (v - 4));
                const int k  = 32 * c + kl + 8 * half;
                a2[c][2 * v]     = h2w[lm * H2_STRIDE + k];
                a2[c][2 * v + 1] = h2w[lm * H2_STRIDE + k + 1];
            }
        }

        // ---- layer 3: out = h2 @ W3 + b3, store f32 (masked NE tail) ----
        #pragma unroll
        for (int j = 0; j < 4; ++j) {                  // 4 N-tiles of 16 (OUT=64)
            const int col = 16 * j + lm;
            v16h b[4];
            #pragma unroll
            for (int c = 0; c < 4; ++c) {
                #pragma unroll
                for (int v = 0; v < 8; ++v) {
                    const int k = 32 * c + 2 * v + (v >= 4 ? 8 : 0) + 8 * half;
                    b[c][2 * v]     = w3t[col * W3T_STRIDE + k];
                    b[c][2 * v + 1] = w3t[col * W3T_STRIDE + k + 1];
                }
            }
            v8f acc = {};
            #pragma unroll
            for (int c = 0; c < 4; ++c) {
                acc = __builtin_amdgcn_wmma_f32_16x16x32_f16(
                          false, a2[c], false, b[c], (short)0, acc, false, false);
            }
            const float bb = b3s[col];
            #pragma unroll
            for (int r = 0; r < 8; ++r) {
                const int m = r + 8 * half;
                const int n = n0 + m;
                if (n < NE) {
                    out[((size_t)p * NE + n) * OUT_D + col] = acc[r] + bb;
                }
            }
        }
    }
}

// ------------------------------------------------------------------
extern "C" void kernel_launch(void* const* d_in, const int* in_sizes, int n_in,
                              void* d_out, int out_size, void* d_ws, size_t ws_size,
                              hipStream_t stream) {
    (void)in_sizes; (void)n_in; (void)out_size; (void)ws_size;
    const int*   zj     = (const int*)d_in[0];
    const int*   zk     = (const int*)d_in[1];
    const float* e_feat = (const float*)d_in[2];
    const float* z_emb  = (const float*)d_in[3];
    const float* W1     = (const float*)d_in[4];
    const float* b1     = (const float*)d_in[5];
    const float* W2     = (const float*)d_in[6];
    const float* b2     = (const float*)d_in[7];
    const float* W3     = (const float*)d_in[8];
    const float* b3     = (const float*)d_in[9];
    float* out = (float*)d_out;

    float* pp = (float*)d_ws;                         // P*HID f32 = 2 MB
    float* ep = pp + (size_t)P_N * HID;               // NE*HID f32 = 100 KB

    pp_kernel<<<P_N, 128, 0, stream>>>(zj, zk, z_emb, W1, pp);
    ep_kernel<<<NE, 128, 0, stream>>>(e_feat, W1, ep);
    mlp_kernel<<<P_N, 256, 0, stream>>>(pp, ep, W2, b2, W3, b3, b1, out);
}
